// MultiHeadAttn_59296318488687
// MI455X (gfx1250) — matrix operand order repaired and checked by reading, not compiled, not run on hardware
//
#include <hip/hip_runtime.h>

// ---------------- CDNA5 WMMA types ----------------
typedef __bf16 v16bf __attribute__((ext_vector_type(16)));
typedef __bf16 v8bf  __attribute__((ext_vector_type(8)));
typedef float  v8f   __attribute__((ext_vector_type(8)));

#define CC   512
#define NN   4096          // H*W
#define BB   8
#define NH   8
#define DD   64
#define GG   32

// D = A(16x32) * B(32x16) + C, bf16 in / f32 out
__device__ __forceinline__ v8f wmma_bf16(v16bf a, v16bf b, v8f c) {
  return __builtin_amdgcn_wmma_f32_16x16x32_bf16(false, a, false, b, (short)0, c, false, false);
}

// ---- CDNA5 async global->LDS copy (ASYNCcnt-tracked) ----
__device__ __forceinline__ void async_ld_b128(unsigned lds_byte_off, const __bf16* g) {
  asm volatile("global_load_async_to_lds_b128 %0, %1, off" :: "v"(lds_byte_off), "v"(g) : "memory");
}
__device__ __forceinline__ void wait_async0() {
  asm volatile("s_wait_asynccnt 0x0" ::: "memory");
}
__device__ __forceinline__ unsigned lds_off(const void* p) {
  return (unsigned)(size_t)p;   // flat LDS address: low 32 bits = LDS offset
}

// A fragment 16x32 (row-major source, ld in elements).
__device__ __forceinline__ v16bf load_frag_a(const __bf16* base, int ld, int lane) {
  int row  = lane & 15;
  int koff = (lane >> 4) << 3;
  union { v16bf v; v8bf h[2]; } u;
  u.h[0] = *(const v8bf*)(base + (size_t)row * ld + koff);
  u.h[1] = *(const v8bf*)(base + (size_t)row * ld + 16 + koff);
  return u.v;
}

// B fragment 32x16 loaded from TRANSPOSED source Bt[N][K] (ld in elements).
__device__ __forceinline__ v16bf load_frag_bt(const __bf16* baseT, int ld, int lane) {
  int col  = lane & 15;
  int koff = (lane >> 4) << 3;
  union { v16bf v; v8bf h[2]; } u;
  u.h[0] = *(const v8bf*)(baseT + (size_t)col * ld + koff);
  u.h[1] = *(const v8bf*)(baseT + (size_t)col * ld + 16 + koff);
  return u.v;
}

__device__ __forceinline__ float elu1(float v) { return v > 0.f ? v + 1.f : __expf(v); }

// ---------------- Kernel A: GroupNorm -> xnT bf16 [b][n][c] ----------------
__global__ void gn_kernel(const float* __restrict__ x, const float* __restrict__ gamma,
                          const float* __restrict__ beta, __bf16* __restrict__ xnT) {
  int b = blockIdx.x >> 5, g = blockIdx.x & 31;
  int cb = g * 16, tid = threadIdx.x;
  float s = 0.f, ss = 0.f;
  for (int c = 0; c < 16; ++c) {
    const float* xp = x + ((size_t)b * CC + cb + c) * NN;
    #pragma unroll
    for (int j = 0; j < 16; ++j) { float v = xp[tid + j * 256]; s += v; ss += v * v; }
  }
  __shared__ float red[512];
  __shared__ float mv[2];
  red[tid] = s; red[256 + tid] = ss;
  __syncthreads();
  for (int st = 128; st > 0; st >>= 1) {
    if (tid < st) { red[tid] += red[tid + st]; red[256 + tid] += red[256 + tid + st]; }
    __syncthreads();
  }
  if (tid == 0) {
    float mean = red[0] * (1.f / 65536.f);
    float var  = red[256] * (1.f / 65536.f) - mean * mean;
    mv[0] = mean; mv[1] = rsqrtf(var + 1e-5f);
  }
  __syncthreads();
  float mean = mv[0], rstd = mv[1];
  float gw[16], bw[16];
  #pragma unroll
  for (int c = 0; c < 16; ++c) {
    float gm = gamma[cb + c] * rstd;
    gw[c] = gm; bw[c] = beta[cb + c] - mean * gm;
  }
  for (int j = 0; j < 16; ++j) {
    int n = tid + j * 256;
    v8bf b0, b1;
    #pragma unroll
    for (int c = 0; c < 8; ++c)
      b0[c] = (__bf16)(x[((size_t)b * CC + cb + c) * NN + n] * gw[c] + bw[c]);
    #pragma unroll
    for (int c = 0; c < 8; ++c)
      b1[c] = (__bf16)(x[((size_t)b * CC + cb + 8 + c) * NN + n] * gw[8 + c] + bw[8 + c]);
    __bf16* dp = xnT + ((size_t)b * NN + n) * CC + cb;
    *(v8bf*)dp = b0; *(v8bf*)(dp + 8) = b1;
  }
}

// ---------------- Kernel W: weights fp32 -> bf16 ----------------
__global__ void wconv_kernel(const float* __restrict__ qkv_w, const float* __restrict__ proj_w,
                             __bf16* __restrict__ wq, __bf16* __restrict__ wp) {
  int idx = blockIdx.x * 256 + threadIdx.x;
  if (idx < 3 * CC * CC) wq[idx] = (__bf16)qkv_w[idx];
  else {
    int j = idx - 3 * CC * CC;
    if (j < CC * CC) wp[j] = (__bf16)proj_w[j];
  }
}

// ---------------- Kernel B: QKV GEMM + bias + elu1, head-split layouts ----------------
// Block: 128 o x 64 n. Shared B tile (64 n x 32 k) async-staged to LDS, double buffered.
// All four B fragments loaded before the WMMA chain (one DS clause per k-step).
__global__ void qkv_kernel(const __bf16* __restrict__ wq, const __bf16* __restrict__ xnT,
                           const float* __restrict__ qkv_b,
                           __bf16* __restrict__ q, __bf16* __restrict__ kT, __bf16* __restrict__ vT) {
  int b = blockIdx.z, om = blockIdx.y * 128, nm = blockIdx.x * 64;
  int w = threadIdx.x >> 5, lane = threadIdx.x & 31;
  int ob = om + w * 16;
  int hb = (lane >> 4) << 3;
  __shared__ __bf16 bsm[2][64][32];

  int srow = w * 8 + (lane >> 2);
  int scol = (lane & 3) * 8;                       // element offset in row (8 bf16 = 16B)
  const __bf16* gsrc_row = xnT + ((size_t)b * NN + nm + srow) * CC + scol;
  unsigned lbuf0 = lds_off(&bsm[0][srow][scol]);
  unsigned lbuf1 = lds_off(&bsm[1][srow][scol]);

  float biasr[8];
  #pragma unroll
  for (int r = 0; r < 8; ++r) biasr[r] = qkv_b[ob + r + hb];
  v8f acc[4] = {};
  const __bf16* abase = wq + (size_t)ob * CC;

  async_ld_b128(lbuf0, gsrc_row);   // k-chunk 0
  wait_async0();
  __syncthreads();

  #pragma unroll
  for (int j = 0; j < CC / 32; ++j) {
    if (j + 1 < CC / 32)
      async_ld_b128((j & 1) ? lbuf0 : lbuf1, gsrc_row + (j + 1) * 32);
    v16bf a = load_frag_a(abase + j * 32, CC, lane);
    v16bf bm[4];
    #pragma unroll
    for (int t = 0; t < 4; ++t) bm[t] = load_frag_bt(&bsm[j & 1][t * 16][0], 32, lane);
    #pragma unroll
    for (int t = 0; t < 4; ++t) acc[t] = wmma_bf16(a, bm[t], acc[t]);
    wait_async0();
    __syncthreads();
  }

  int sect  = ob >> 9;       // 0:q 1:k 2:v
  int oloc  = ob & 511;
  int h     = oloc >> 6;
  int dtile = oloc & 63;
  if (sect == 0) {
    #pragma unroll
    for (int t = 0; t < 4; ++t) {
      int n = nm + t * 16 + (lane & 15);
      v8bf buf;
      #pragma unroll
      for (int r = 0; r < 8; ++r) buf[r] = (__bf16)elu1(acc[t][r] + biasr[r]);
      *(v8bf*)(q + (((size_t)(b * NH + h)) * NN + n) * DD + dtile + hb) = buf;
    }
  } else {
    __bf16* dst = (sect == 1) ? kT : vT;
    #pragma unroll
    for (int t = 0; t < 4; ++t) {
      int n = nm + t * 16 + (lane & 15);
      #pragma unroll
      for (int r = 0; r < 8; ++r) {
        int d = dtile + r + hb;
        float v = acc[t][r] + biasr[r];
        if (sect == 1) v = elu1(v);
        dst[(((size_t)(b * NH + h)) * DD + d) * NN + n] = (__bf16)v;
      }
    }
  }
}

// ---------------- Kernel C: kv = k^T v  (per b,h) + ksum ----------------
// n-loop unrolled x4; fragment loads batched ahead of the WMMA chain.
__global__ void kv_kernel(const __bf16* __restrict__ kT, const __bf16* __restrict__ vT,
                          __bf16* __restrict__ kvT, float* __restrict__ ksum) {
  int bh = blockIdx.x;
  int w = threadIdx.x >> 5, lane = threadIdx.x & 31;
  int dt = w & 3, et = w >> 2;
  int hb = (lane >> 4) << 3;
  size_t abase = ((size_t)bh * DD + dt * 16) * NN;
  v8f acc[2] = {};
  float ks = 0.f;
  for (int n0 = 0; n0 < NN; n0 += 128) {
    #pragma unroll
    for (int u = 0; u < 4; ++u) {
      int nn = n0 + u * 32;
      union { v16bf v; v8bf h[2]; } ua;
      ua.v = load_frag_a(kT + abase + nn, NN, lane);
      v16bf bm[2];
      #pragma unroll
      for (int t = 0; t < 2; ++t)
        bm[t] = load_frag_bt(vT + ((size_t)bh * DD + et * 32 + t * 16) * NN + nn, NN, lane);
      if (et == 0) {
        #pragma unroll
        for (int i = 0; i < 16; ++i) ks += (float)ua.v[i];
      }
      #pragma unroll
      for (int t = 0; t < 2; ++t) acc[t] = wmma_bf16(ua.v, bm[t], acc[t]);
    }
  }
  if (et == 0) {
    ks += __shfl_xor(ks, 16, 32);
    if (lane < 16) ksum[bh * DD + dt * 16 + lane] = ks;
  }
  #pragma unroll
  for (int t = 0; t < 2; ++t) {
    v8bf buf;
    #pragma unroll
    for (int r = 0; r < 8; ++r) buf[r] = (__bf16)acc[t][r];
    int e = et * 32 + t * 16 + (lane & 15);
    *(v8bf*)(kvT + ((size_t)bh * DD + e) * DD + dt * 16 + hb) = buf;  // kvT[e][d]
  }
}

// ---------------- Kernel D: out = q·kv ; denom = q·ksum ; attnT[b][n][c] ----------------
// Whole kvT (64x64 bf16 = 8KB) async-staged into LDS once per block.
__global__ void attn_out_kernel(const __bf16* __restrict__ q, const __bf16* __restrict__ kvT,
                                const float* __restrict__ ksum, __bf16* __restrict__ attnT) {
  int bh = blockIdx.y, b = bh >> 3, h = bh & 7;
  int w = threadIdx.x >> 5, lane = threadIdx.x & 31;
  __shared__ __bf16 skv[64][64];
  __shared__ float sks[64];
  __shared__ float sden[8][16];
  {  // async stage: 8KB = 256 threads x 2 x 16B
    const __bf16* gkv = kvT + (size_t)bh * DD * DD + threadIdx.x * 16;
    unsigned lo = lds_off(&skv[0][0]) + threadIdx.x * 32;
    async_ld_b128(lo, gkv);
    async_ld_b128(lo + 16, gkv + DD * DD / 2);
  }
  if (threadIdx.x < 64) sks[threadIdx.x] = ksum[bh * DD + threadIdx.x];
  wait_async0();
  __syncthreads();

  int nb = blockIdx.x * 128 + w * 16;
  size_t qbase = ((size_t)bh * NN + nb) * DD;
  int hb = (lane >> 4) << 3;
  v8f acc[4] = {};
  float dn = 0.f;
  #pragma unroll
  for (int k0 = 0; k0 < DD; k0 += 32) {
    union { v16bf v; v8bf h[2]; } ua;
    ua.v = load_frag_a(q + qbase + k0, DD, lane);
    v16bf bm[4];
    #pragma unroll
    for (int t = 0; t < 4; ++t) bm[t] = load_frag_bt(&skv[t * 16][0] + k0, DD, lane);
    int kb = k0 + hb;
    #pragma unroll
    for (int i = 0; i < 8; ++i) dn += (float)ua.v[i] * sks[kb + i];
    #pragma unroll
    for (int i = 0; i < 8; ++i) dn += (float)ua.v[8 + i] * sks[kb + 16 + i];
    #pragma unroll
    for (int t = 0; t < 4; ++t) acc[t] = wmma_bf16(ua.v, bm[t], acc[t]);
  }
  dn += __shfl_xor(dn, 16, 32);
  if (lane < 16) sden[w][lane] = dn;
  __syncthreads();
  #pragma unroll
  for (int t = 0; t < 4; ++t) {
    int c = h * DD + t * 16 + (lane & 15);
    #pragma unroll
    for (int r = 0; r < 8; ++r) {
      int m = r + hb;
      float den = sden[w][m] + 1e-6f;
      attnT[((size_t)b * NN + nb + m) * CC + c] = (__bf16)(acc[t][r] / den);
    }
  }
}

// ---------------- Kernel E: proj GEMM + bias + residual -> fp32 out ----------------
__global__ void proj_kernel(const __bf16* __restrict__ wp, const __bf16* __restrict__ attnT,
                            const float* __restrict__ proj_b, const float* __restrict__ x,
                            float* __restrict__ out) {
  int b = blockIdx.z, om = blockIdx.y * 128, nm = blockIdx.x * 64;
  int w = threadIdx.x >> 5, lane = threadIdx.x & 31;
  int ob = om + w * 16;
  int hb = (lane >> 4) << 3;
  __shared__ __bf16 bsm[2][64][32];

  int srow = w * 8 + (lane >> 2);
  int scol = (lane & 3) * 8;
  const __bf16* gsrc_row = attnT + ((size_t)b * NN + nm + srow) * CC + scol;
  unsigned lbuf0 = lds_off(&bsm[0][srow][scol]);
  unsigned lbuf1 = lds_off(&bsm[1][srow][scol]);

  float biasr[8];
  #pragma unroll
  for (int r = 0; r < 8; ++r) biasr[r] = proj_b[ob + r + hb];
  v8f acc[4] = {};
  const __bf16* abase = wp + (size_t)ob * CC;

  async_ld_b128(lbuf0, gsrc_row);
  wait_async0();
  __syncthreads();

  #pragma unroll
  for (int j = 0; j < CC / 32; ++j) {
    if (j + 1 < CC / 32)
      async_ld_b128((j & 1) ? lbuf0 : lbuf1, gsrc_row + (j + 1) * 32);
    v16bf a = load_frag_a(abase + j * 32, CC, lane);
    v16bf bm[4];
    #pragma unroll
    for (int t = 0; t < 4; ++t) bm[t] = load_frag_bt(&bsm[j & 1][t * 16][0], 32, lane);
    #pragma unroll
    for (int t = 0; t < 4; ++t) acc[t] = wmma_bf16(a, bm[t], acc[t]);
    wait_async0();
    __syncthreads();
  }

  #pragma unroll
  for (int t = 0; t < 4; ++t) {
    int n = nm + t * 16 + (lane & 15);
    #pragma unroll
    for (int r = 0; r < 8; ++r) {
      int o = ob + r + hb;
      size_t idx = ((size_t)b * CC + o) * NN + n;
      out[idx] = acc[t][r] + biasr[r] + x[idx];
    }
  }
}

// ---------------- launcher ----------------
extern "C" void kernel_launch(void* const* d_in, const int* in_sizes, int n_in,
                              void* d_out, int out_size, void* d_ws, size_t ws_size,
                              hipStream_t stream) {
  const float* x     = (const float*)d_in[0];
  const float* gamma = (const float*)d_in[1];
  const float* beta  = (const float*)d_in[2];
  const float* qkv_w = (const float*)d_in[3];
  const float* qkv_b = (const float*)d_in[4];
  const float* projw = (const float*)d_in[5];
  const float* projb = (const float*)d_in[6];
  float* out = (float*)d_out;

  char* ws = (char*)d_ws;
  const size_t SZ_BNC = (size_t)BB * NN * CC * sizeof(__bf16);   // 32 MB
  __bf16* xnT  = (__bf16*)(ws);                    // also reused as attnT after QKV
  __bf16* qb   = (__bf16*)(ws + SZ_BNC);
  __bf16* kT   = (__bf16*)(ws + 2 * SZ_BNC);
  __bf16* vT   = (__bf16*)(ws + 3 * SZ_BNC);
  __bf16* wq   = (__bf16*)(ws + 4 * SZ_BNC);
  __bf16* wp   = (__bf16*)(ws + 4 * SZ_BNC + 3 * CC * CC * sizeof(__bf16));
  __bf16* kvT  = (__bf16*)(ws + 4 * SZ_BNC + 4 * CC * CC * sizeof(__bf16));
  float*  ksum = (float*) (ws + 4 * SZ_BNC + 4 * CC * CC * sizeof(__bf16)
                              + (size_t)BB * NH * DD * DD * sizeof(__bf16));

  gn_kernel<<<BB * GG, 256, 0, stream>>>(x, gamma, beta, xnT);
  wconv_kernel<<<(4 * CC * CC) / 256, 256, 0, stream>>>(qkv_w, projw, wq, wp);
  qkv_kernel<<<dim3(NN / 64, (3 * CC) / 128, BB), 256, 0, stream>>>(wq, xnT, qkv_b, qb, kT, vT);
  kv_kernel<<<BB * NH, 256, 0, stream>>>(kT, vT, kvT, ksum);
  __bf16* attnT = xnT;  // reuse
  attn_out_kernel<<<dim3(NN / 128, BB * NH), 256, 0, stream>>>(qb, kvT, ksum, attnT);
  proj_kernel<<<dim3(NN / 64, CC / 128, BB), 256, 0, stream>>>(wp, attnT, projb, x, out);
}
